// MiniCPMSparseFlashAttention2_46909632806935
// MI455X (gfx1250) — matrix operand order repaired and checked by reading, not compiled, not run on hardware
//
#include <hip/hip_runtime.h>

// MiniCPM sparse flash attention for gfx1250 (MI455X), wave32 + WMMA bf16.
// Pipeline:
//   (0a) prep: K -> bf16 row-major, V -> bf16 transposed [kv][d][s] (LDS tile)
//   (0b) qconv: Q -> bf16 row-major (uint4-vectorized)
//   (1)  mean-pool K -> bf16 compressed keys
//   (2)  WMMA compressed scores + masked softmax + block max-pool + top-4
//   (3)  WMMA block-sparse attention over gathered blocks (P via LDS transpose)

typedef __attribute__((ext_vector_type(16))) __bf16 v16bf;
typedef __attribute__((ext_vector_type(8)))  float  v8f;

namespace {
constexpr int S = 2048, HQ = 16, HKV = 2, D = 128, G = 8;
constexpr int KER = 32, STRIDE = 16, BLOCK = 64, TOPK = 4;
constexpr int C  = (S - KER) / STRIDE + 1;   // 127 compressed keys
constexpr int NB = S / BLOCK;                // 32 KV blocks
constexpr float SCALE = 0.08838834764831845f; // 1/sqrt(128)
constexpr float NEGF  = -1e30f;

constexpr size_t OFF_TOPK = 65536;                              // after kcmp
constexpr size_t OFF_KB   = 131072;
constexpr size_t OFF_VT   = OFF_KB + (size_t)S * HKV * D * 2;   // +1 MB
constexpr size_t OFF_QB   = OFF_VT + (size_t)S * HKV * D * 2;   // +1 MB
}

union BF16x16 {
  v16bf v;
  unsigned short h[16];
  unsigned int u[8];
  uint4 q[2];
};

static __device__ __forceinline__ unsigned short f2bf(float f) {
  unsigned u = __builtin_bit_cast(unsigned, f);
  unsigned r = u + 0x7FFFu + ((u >> 16) & 1u);   // round-to-nearest-even
  return (unsigned short)(r >> 16);
}

static __device__ __forceinline__ unsigned int pk2bf(float a, float b) {
  return (unsigned int)f2bf(a) | ((unsigned int)f2bf(b) << 16);
}

static __device__ __forceinline__ v8f wmma_bf16(v16bf a, v16bf b, v8f c) {
  return __builtin_amdgcn_wmma_f32_16x16x32_bf16(false, a, false, b, (short)0, c,
                                                 false, false);
}

// ---------------------------------------------------------------------------
// Kernel 0a: K -> bf16 row-major kb16[kv][s][d]; V -> bf16 transposed
// vt16[kv][d][s] via padded LDS tile (both directions coalesced).
// Grid (S/16, D/16, HKV), 256 threads (16x16 tile).
// ---------------------------------------------------------------------------
__global__ __launch_bounds__(256) void nsa_prep(const float* __restrict__ kmat,
                                                const float* __restrict__ vmat,
                                                unsigned short* __restrict__ kb16,
                                                unsigned short* __restrict__ vt16) {
  __shared__ unsigned short tile[16][17];     // pad to dodge bank conflicts
  const int tx = threadIdx.x & 15;            // d within tile
  const int ty = threadIdx.x >> 4;            // s within tile
  const int s0 = blockIdx.x * 16, d0 = blockIdx.y * 16, kv = blockIdx.z;
  const int s = s0 + ty, d = d0 + tx;

  const size_t src = ((size_t)s * HKV + kv) * D + d;
  kb16[((size_t)kv * S + s) * D + d] = f2bf(kmat[src]);
  tile[ty][tx] = f2bf(vmat[src]);
  __syncthreads();
  // write transposed: row = d0+ty (d), col = s0+tx (s); 32B per 16 lanes
  vt16[((size_t)kv * D + d0 + ty) * S + s0 + tx] = tile[tx][ty];
}

// ---------------------------------------------------------------------------
// Kernel 0b: Q -> bf16 row-major (8 floats -> one uint4 store per thread).
// Grid (S*HQ*D / 2048), 256 threads.
// ---------------------------------------------------------------------------
__global__ __launch_bounds__(256) void nsa_qconv(const float* __restrict__ q,
                                                 unsigned short* __restrict__ qb16) {
  const size_t i = ((size_t)blockIdx.x * 256 + threadIdx.x) * 8;
  float4 a = *(const float4*)(q + i);
  float4 b = *(const float4*)(q + i + 4);
  uint4 o;
  o.x = pk2bf(a.x, a.y);
  o.y = pk2bf(a.z, a.w);
  o.z = pk2bf(b.x, b.y);
  o.w = pk2bf(b.z, b.w);
  *(uint4*)(qb16 + i) = o;
}

// ---------------------------------------------------------------------------
// Kernel 1: compressed (mean-pooled) keys, stored bf16 as [kv][c][d].
// ---------------------------------------------------------------------------
__global__ __launch_bounds__(128) void nsa_kcmp(const float* __restrict__ kmat,
                                                unsigned short* __restrict__ kcmp) {
  const int c = blockIdx.x, kv = blockIdx.y, d = threadIdx.x;
  const float* base = kmat + ((size_t)(c * STRIDE) * HKV + kv) * D + d;
  float s = 0.f;
  #pragma unroll
  for (int j = 0; j < KER; ++j) s += base[(size_t)j * HKV * D];
  kcmp[((size_t)kv * C + c) * D + d] = f2bf(s * (1.f / KER));
}

// ---------------------------------------------------------------------------
// Kernel 2: compressed scores + softmax + group-sum + block max-pool + top-k.
// Grid (S/16, HKV), 256 threads (8 waves; wave w = group head w).
// ---------------------------------------------------------------------------
__global__ __launch_bounds__(256) void nsa_score_topk(const unsigned short* __restrict__ qb16,
                                                      const unsigned short* __restrict__ kcmp,
                                                      int* __restrict__ topk) {
  __shared__ float score[16][128];   // per-query summed-over-group probs
  const int tid = threadIdx.x, wave = tid >> 5, lane = tid & 31;
  const int Mloc = lane & 15, hi = lane >> 4;
  const int s0 = blockIdx.x * 16, kv = blockIdx.y;

  for (int i = tid; i < 16 * 128; i += 256) (&score[0][0])[i] = 0.f;
  __syncthreads();

  const int head = kv * G + wave;

  // A operand: 16 query rows s0..s0+15, K = D (4 chunks of 32), bf16 direct.
  BF16x16 A[4];
  {
    const unsigned short* qr = qb16 + ((size_t)(s0 + Mloc) * HQ + head) * D;
    #pragma unroll
    for (int kc = 0; kc < 4; ++kc) {
      const unsigned short* p = qr + kc * 32 + hi * 16;
      A[kc].q[0] = *(const uint4*)p;         // K = dlo..dlo+7
      A[kc].q[1] = *(const uint4*)(p + 8);   // K = dlo+8..dlo+15
    }
  }

  // Logits: 8 column tiles of 16 compressed keys.
  v8f lg[8];
  #pragma unroll
  for (int t = 0; t < 8; ++t) {
    const int c = t * 16 + Mloc;          // B lane column
    const bool cv = c < C;
    const unsigned short* kr = kcmp + ((size_t)kv * C + (cv ? c : 0)) * D;
    v8f acc = {0.f, 0.f, 0.f, 0.f, 0.f, 0.f, 0.f, 0.f};
    #pragma unroll
    for (int kc = 0; kc < 4; ++kc) {
      BF16x16 b;
      if (cv) {
        const unsigned short* p = kr + kc * 32 + hi * 16;
        b.q[0] = *(const uint4*)p;
        b.q[1] = *(const uint4*)(p + 8);
      } else {
        b.q[0] = make_uint4(0, 0, 0, 0);
        b.q[1] = make_uint4(0, 0, 0, 0);
      }
      acc = wmma_bf16(A[kc].v, b.v, acc);
    }
    lg[t] = acc;
  }

  // Masked softmax per query row (D-layout: row = r + hi*8, col = t*16+Mloc),
  // then accumulate probs over group heads into LDS.
  #pragma unroll
  for (int r = 0; r < 8; ++r) {
    const int srow = s0 + r + hi * 8;
    float m = NEGF;
    #pragma unroll
    for (int t = 0; t < 8; ++t) {
      const int c = t * 16 + Mloc;
      float x = lg[t][r] * SCALE;
      const bool vis = (c < C) && (c * STRIDE + KER - 1 <= srow);
      x = vis ? x : NEGF;
      lg[t][r] = x;
      m = fmaxf(m, x);
    }
    #pragma unroll
    for (int off = 1; off <= 8; off <<= 1) m = fmaxf(m, __shfl_xor(m, off, 32));
    float sum = 0.f;
    #pragma unroll
    for (int t = 0; t < 8; ++t) {
      const float x = lg[t][r];
      const float e = (x <= 0.5f * NEGF) ? 0.f : __expf(x - m);
      lg[t][r] = e;
      sum += e;
    }
    #pragma unroll
    for (int off = 1; off <= 8; off <<= 1) sum += __shfl_xor(sum, off, 32);
    const float inv = sum > 0.f ? 1.f / sum : 0.f;
    #pragma unroll
    for (int t = 0; t < 8; ++t) {
      const int c = t * 16 + Mloc;
      if (c < C) atomicAdd(&score[r + hi * 8][c], lg[t][r] * inv);
    }
  }
  __syncthreads();

  // Block max-pool + forced/causal overrides + top-4 select/sort/remap.
  if (tid < 16) {
    const int s = s0 + tid;
    const int qb = s >> 6;
    float blk[NB];
    for (int b = 0; b < NB; ++b) {
      int clo = 4 * b - 1; if (clo < 0) clo = 0;
      int chi = 4 * b + 3; if (chi > C - 1) chi = C - 1;
      float mx = NEGF;
      for (int c = clo; c <= chi; ++c) mx = fmaxf(mx, score[tid][c]);
      if (b > qb)                     mx = NEGF;      // causal (overrides forced)
      else if (b < 1 || b >= qb - 1)  mx = 1e9f;      // init + local forced
      blk[b] = mx;
    }
    int idx[TOPK];
    #pragma unroll
    for (int j = 0; j < TOPK; ++j) {
      int bi = 0; float bv = blk[0];
      for (int b = 1; b < NB; ++b) if (blk[b] > bv) { bv = blk[b]; bi = b; }
      idx[j] = bi;
      blk[bi] = -3.0e38f;
    }
    #pragma unroll
    for (int a = 0; a < TOPK - 1; ++a)
      #pragma unroll
      for (int b2 = a + 1; b2 < TOPK; ++b2)
        if (idx[b2] < idx[a]) { const int t2 = idx[a]; idx[a] = idx[b2]; idx[b2] = t2; }
    #pragma unroll
    for (int j = 0; j < TOPK; ++j)
      topk[((size_t)kv * S + s) * TOPK + j] = (idx[j] >= qb) ? -1 : idx[j];
  }
}

// ---------------------------------------------------------------------------
// Kernel 3: block-sparse attention. Grid (S/4, HKV), 128 threads (4 waves),
// one wave per (kv, s). All selected blocks are strictly past -> no intra-
// block causal mask. P round-trips through LDS (D-layout -> A-layout).
// Q/K/V all read as pre-converted bf16: qb16/kb16 row-major, vt16 transposed.
// ---------------------------------------------------------------------------
__global__ __launch_bounds__(128) void nsa_sparse_attn(const unsigned short* __restrict__ qb16,
                                                       const unsigned short* __restrict__ kb16,
                                                       const unsigned short* __restrict__ vt16,
                                                       const int* __restrict__ topk,
                                                       float* __restrict__ out) {
  __shared__ unsigned short Pall[4 * 16 * 256];   // 32 KB: per-wave P (bf16)
  const int tid = threadIdx.x, wave = tid >> 5, lane = tid & 31;
  const int Mloc = lane & 15, hi = lane >> 4;
  const int kv = blockIdx.y;
  const int s = blockIdx.x * 4 + wave;
  unsigned short* P = Pall + wave * 16 * 256;

  int bidx[TOPK];
  #pragma unroll
  for (int j = 0; j < TOPK; ++j) bidx[j] = topk[((size_t)kv * S + s) * TOPK + j];

  // A operand: rows = 8 group heads of query s, rows 8..15 zero.
  BF16x16 A[4];
  {
    const bool hv = Mloc < G;
    const unsigned short* qr = qb16 + ((size_t)s * HQ + kv * G + (hv ? Mloc : 0)) * D;
    #pragma unroll
    for (int kc = 0; kc < 4; ++kc) {
      if (hv) {
        const unsigned short* p = qr + kc * 32 + hi * 16;
        A[kc].q[0] = *(const uint4*)p;
        A[kc].q[1] = *(const uint4*)(p + 8);
      } else {
        A[kc].q[0] = make_uint4(0, 0, 0, 0);
        A[kc].q[1] = make_uint4(0, 0, 0, 0);
      }
    }
  }

  // QK^T over 4 gathered blocks = 256 key slots = 16 N-tiles.
  v8f lg[16];
  #pragma unroll
  for (int t = 0; t < 16; ++t) {
    const int bb = bidx[t >> 2];
    const int key = (bb >= 0 ? bb : 0) * BLOCK + (t & 3) * 16 + Mloc;  // B column
    const unsigned short* kr = kb16 + ((size_t)kv * S + key) * D;
    v8f acc = {0.f, 0.f, 0.f, 0.f, 0.f, 0.f, 0.f, 0.f};
    #pragma unroll
    for (int kc = 0; kc < 4; ++kc) {
      BF16x16 b;
      const unsigned short* p = kr + kc * 32 + hi * 16;
      b.q[0] = *(const uint4*)p;
      b.q[1] = *(const uint4*)(p + 8);
      acc = wmma_bf16(A[kc].v, b.v, acc);
    }
    lg[t] = acc;
  }

  // Softmax per head row; write unnormalized exp(P) bf16 to LDS.
  float inv[8];
  #pragma unroll
  for (int r = 0; r < 8; ++r) {
    float m = NEGF;
    #pragma unroll
    for (int t = 0; t < 16; ++t) {
      const float x = (bidx[t >> 2] >= 0) ? lg[t][r] * SCALE : NEGF;
      lg[t][r] = x;
      m = fmaxf(m, x);
    }
    #pragma unroll
    for (int off = 1; off <= 8; off <<= 1) m = fmaxf(m, __shfl_xor(m, off, 32));
    float sum = 0.f;
    #pragma unroll
    for (int t = 0; t < 16; ++t) {
      const float x = lg[t][r];
      const float e = (x <= 0.5f * NEGF) ? 0.f : __expf(x - m);
      lg[t][r] = e;
      sum += e;
    }
    #pragma unroll
    for (int off = 1; off <= 8; off <<= 1) sum += __shfl_xor(sum, off, 32);
    inv[r] = sum > 0.f ? 1.f / sum : 0.f;
    const int row = r + hi * 8;
    #pragma unroll
    for (int t = 0; t < 16; ++t) P[row * 256 + t * 16 + Mloc] = f2bf(lg[t][r]);
  }

  // PV: O[16 heads x 128 d] over 256 key slots (8 K-chunks of 32).
  v8f o[8];
  #pragma unroll
  for (int n = 0; n < 8; ++n)
    #pragma unroll
    for (int i = 0; i < 8; ++i) o[n][i] = 0.f;

  #pragma unroll
  for (int kc = 0; kc < 8; ++kc) {
    BF16x16 pa;   // A operand from LDS (same-wave LDS ops are in order)
    const unsigned short* pr = P + Mloc * 256 + kc * 32 + hi * 16;
    pa.q[0] = *(const uint4*)pr;
    pa.q[1] = *(const uint4*)(pr + 8);
    const int bb = bidx[kc >> 1];
    const int base = (bb >= 0 ? bb : 0) * BLOCK + (kc & 1) * 32 + hi * 16;
    #pragma unroll
    for (int n = 0; n < 8; ++n) {
      BF16x16 vb;   // B operand: col = d, K = 16 consecutive keys (contiguous!)
      const int d = n * 16 + Mloc;
      const unsigned short* p = vt16 + ((size_t)kv * D + d) * S + base;
      vb.q[0] = *(const uint4*)p;
      vb.q[1] = *(const uint4*)(p + 8);
      o[n] = wmma_bf16(pa.v, vb.v, o[n]);
    }
  }

  // D-layout rows 0..7 (lanes 0..15) are the valid heads; normalize and store.
  if (lane < 16) {
    #pragma unroll
    for (int r = 0; r < 8; ++r) {
      float* orow = out + ((size_t)s * HQ + kv * G + r) * D;
      #pragma unroll
      for (int n = 0; n < 8; ++n) orow[n * 16 + Mloc] = o[n][r] * inv[r];
    }
  }
}

// ---------------------------------------------------------------------------
extern "C" void kernel_launch(void* const* d_in, const int* in_sizes, int n_in,
                              void* d_out, int out_size, void* d_ws, size_t ws_size,
                              hipStream_t stream) {
  (void)in_sizes; (void)n_in; (void)out_size; (void)ws_size;
  const float* q = (const float*)d_in[0];
  const float* k = (const float*)d_in[1];
  const float* v = (const float*)d_in[2];
  float* out = (float*)d_out;

  char* ws = (char*)d_ws;
  unsigned short* kcmp = (unsigned short*)ws;                   // 65,024 B used
  int*            topk = (int*)(ws + OFF_TOPK);                 // 65,536 B
  unsigned short* kb16 = (unsigned short*)(ws + OFF_KB);        // 1 MB
  unsigned short* vt16 = (unsigned short*)(ws + OFF_VT);        // 1 MB
  unsigned short* qb16 = (unsigned short*)(ws + OFF_QB);        // 8 MB

  nsa_prep<<<dim3(S / 16, D / 16, HKV), 256, 0, stream>>>(k, v, kb16, vt16);
  nsa_qconv<<<dim3((S * HQ * D) / (256 * 8)), 256, 0, stream>>>(q, qb16);
  nsa_kcmp<<<dim3(C, HKV), 128, 0, stream>>>(k, kcmp);
  nsa_score_topk<<<dim3(S / 16, HKV), 256, 0, stream>>>(qb16, kcmp, topk);
  nsa_sparse_attn<<<dim3(S / 4, HKV), 128, 0, stream>>>(qb16, kb16, vt16, topk, out);
}
